// EmotionMoE_35820027249007
// MI455X (gfx1250) — matrix operand order repaired and checked by reading, not compiled, not run on hardware
//
#include <hip/hip_runtime.h>
#include <stddef.h>

// ---------------------------------------------------------------------------
// EmotionMoE for MI455X (gfx1250, wave32, WMMA).
// Heavy compute: 22 identical MLP stacks 3072->2048->1024->512 (ReLU) over
// 1024 pair rows, done as f16 WMMA GEMMs (v_wmma_f32_16x16x32_f16) with f32
// accumulation. A tiles are copied global->LDS with the CDNA5 async-to-LDS
// path (GLOBAL_LOAD_ASYNC_TO_LDS_B128, ASYNCcnt); weights are fp32 in memory
// and converted to f16 while staged into LDS (double-buffered ping-pong).
// Everything else is small fused elementwise/reduction kernels.
// ---------------------------------------------------------------------------

typedef __attribute__((ext_vector_type(16))) _Float16 v16h;
typedef __attribute__((ext_vector_type(8)))  float    v8f;

#define BM 128
#define BN 128
#define BK 32
#define LDS_STRIDE 40   // halves; 80B row stride: 16B-aligned, bank-conflict-free

union FragU { v16h v; unsigned u[8]; };
union PackU { _Float16 h[2]; unsigned u; };

__device__ __forceinline__ void async_copy_b128(unsigned ldsOff, const void* gaddr)
{
    // CDNA5 async global->LDS copy: no data VGPRs, tracked by ASYNCcnt.
    asm volatile("global_load_async_to_lds_b128 %0, %1, off"
                 :: "v"(ldsOff), "v"(gaddr) : "memory");
}
__device__ __forceinline__ void async_wait0()
{
    asm volatile("s_wait_asynccnt 0" ::: "memory");
}

// ---------------------------------------------------------------------------
// Tiled GEMM: C[z] = relu(A[z] @ f16(W[z]) + bias[z]), C in f16.
// A: f16 row-major (M x K), W: f32 row-major (K x N), C: f16 row-major (M x N).
// Block 256 thr = 8 waves; wave grid 2(M) x 4(N); wave tile 64x32 = 4x2 WMMA.
// Double-buffered LDS: async-copy A(k+1), fetch W(k+1) into regs -> WMMA on k
// -> convert+store W(k+1) -> s_wait_asynccnt 0 -> barrier.
// ---------------------------------------------------------------------------
__global__ void __launch_bounds__(256, 1)
gemm_wmma_f16(const _Float16* __restrict__ A, long long strideA,
              const float* __restrict__ W, long long strideW,
              const float* __restrict__ bias, long long strideBias,
              _Float16* __restrict__ C, long long strideC,
              int M, int N, int K)
{
    __shared__ __align__(16) _Float16 ldsA[2 * BM * LDS_STRIDE];
    __shared__ __align__(16) _Float16 ldsB[2 * BN * LDS_STRIDE]; // transposed [n][k]

    const int z = blockIdx.z;
    A    += z * strideA;
    W    += z * strideW;
    bias += z * strideBias;
    C    += z * strideC;

    const int tid  = threadIdx.x;
    const int lane = tid & 31;
    const int wave = tid >> 5;
    const int wm   = wave & 1;   // wave row  (0..1) -> 64 rows each
    const int wn   = wave >> 1;  // wave col  (0..3) -> 32 cols each
    const int hlf  = lane >> 4;  // lane half (0..1)
    const int l15  = lane & 15;

    const int blockM = blockIdx.y * BM;
    const int blockN = blockIdx.x * BN;

    // staging indices (fixed per thread)
    const int ar0 = tid >> 2;              // A rows for the 2 chunks
    const int akq = (tid & 3) * 8;         // A k-offset (halves)
    const int kp  = (tid >> 4) * 2;        // W k row pair: kp, kp+1
    const int n8  = (tid & 15) * 8;        // W col block of 8

    const v8f vzero = {0.f,0.f,0.f,0.f,0.f,0.f,0.f,0.f};
    v8f acc[4][2];
#pragma unroll
    for (int mi = 0; mi < 4; ++mi)
#pragma unroll
        for (int ni = 0; ni < 2; ++ni) acc[mi][ni] = vzero;

    const int kTiles = K >> 5;

    float4 bReg[4];

    // ---- prologue: stage tile 0 into buffer 0 --------------------------
    {
        const int k0 = 0;
        // A: async global -> LDS (flat addr low 32 bits == LDS byte offset)
        async_copy_b128((unsigned)(size_t)&ldsA[ar0 * LDS_STRIDE + akq],
                        A + (size_t)(blockM + ar0) * K + k0 + akq);
        async_copy_b128((unsigned)(size_t)&ldsA[(ar0 + 64) * LDS_STRIDE + akq],
                        A + (size_t)(blockM + ar0 + 64) * K + k0 + akq);
        // W: fp32 load, f16 convert, transposed store
        const float* w0 = W + (size_t)(k0 + kp) * N + blockN + n8;
        bReg[0] = reinterpret_cast<const float4*>(w0)[0];
        bReg[1] = reinterpret_cast<const float4*>(w0)[1];
        bReg[2] = reinterpret_cast<const float4*>(w0 + N)[0];
        bReg[3] = reinterpret_cast<const float4*>(w0 + N)[1];
        const float* r0 = reinterpret_cast<const float*>(&bReg[0]);
        const float* r1 = reinterpret_cast<const float*>(&bReg[2]);
#pragma unroll
        for (int jj = 0; jj < 8; ++jj) {
            PackU pk;
            pk.h[0] = (_Float16)r0[jj];
            pk.h[1] = (_Float16)r1[jj];
            *reinterpret_cast<unsigned*>(&ldsB[(n8 + jj) * LDS_STRIDE + kp]) = pk.u;
        }
    }
    async_wait0();
    __syncthreads();

    for (int kt = 0; kt < kTiles; ++kt) {
        const _Float16* curA = ldsA + (kt & 1) * (BM * LDS_STRIDE);
        const _Float16* curB = ldsB + (kt & 1) * (BN * LDS_STRIDE);
        _Float16* nxtA = ldsA + ((kt + 1) & 1) * (BM * LDS_STRIDE);
        _Float16* nxtB = ldsB + ((kt + 1) & 1) * (BN * LDS_STRIDE);
        const bool hasNext = (kt + 1 < kTiles);

        // ---- issue next-tile staging (overlaps the WMMAs below)
        if (hasNext) {
            const int k0 = (kt + 1) * BK;
            // A tile: fire-and-forget async copies into the other buffer
            async_copy_b128((unsigned)(size_t)&nxtA[ar0 * LDS_STRIDE + akq],
                            A + (size_t)(blockM + ar0) * K + k0 + akq);
            async_copy_b128((unsigned)(size_t)&nxtA[(ar0 + 64) * LDS_STRIDE + akq],
                            A + (size_t)(blockM + ar0 + 64) * K + k0 + akq);
            // W tile: fp32 loads into regs (converted + stored after WMMAs)
            const float* w0 = W + (size_t)(k0 + kp) * N + blockN + n8;
            bReg[0] = reinterpret_cast<const float4*>(w0)[0];
            bReg[1] = reinterpret_cast<const float4*>(w0)[1];
            bReg[2] = reinterpret_cast<const float4*>(w0 + N)[0];
            bReg[3] = reinterpret_cast<const float4*>(w0 + N)[1];
            if (kt + 2 < kTiles) // prefetch k+2 weights (global_prefetch_b8)
                __builtin_prefetch(w0 + (size_t)BK * N, 0, 1);
        }

        // ---- build fragments per the ISA 16-bit A/B layouts (wave32)
        v16h afrag[4], bfrag[2];
#pragma unroll
        for (int mi = 0; mi < 4; ++mi) {
            int row = wm * 64 + mi * 16 + l15;
            const unsigned* base = reinterpret_cast<const unsigned*>(curA + row * LDS_STRIDE);
            FragU f;
#pragma unroll
            for (int i = 0; i < 8; ++i) {
                int kk = ((i < 4) ? (2 * i) : (16 + 2 * (i - 4))) + hlf * 8;
                f.u[i] = base[kk >> 1];
            }
            afrag[mi] = f.v;
        }
#pragma unroll
        for (int ni = 0; ni < 2; ++ni) {
            int col = wn * 32 + ni * 16 + l15;
            const unsigned* base = reinterpret_cast<const unsigned*>(curB + col * LDS_STRIDE);
            FragU f;
#pragma unroll
            for (int i = 0; i < 8; ++i) {
                int kk = ((i < 4) ? (2 * i) : (16 + 2 * (i - 4))) + hlf * 8;
                f.u[i] = base[kk >> 1];
            }
            bfrag[ni] = f.v;
        }

#pragma unroll
        for (int mi = 0; mi < 4; ++mi)
#pragma unroll
            for (int ni = 0; ni < 2; ++ni)
                acc[mi][ni] = __builtin_amdgcn_wmma_f32_16x16x32_f16(
                    false, afrag[mi], false, bfrag[ni],
                    (short)0, acc[mi][ni], false, false);

        // ---- convert + commit next W tile into the other LDS buffer
        if (hasNext) {
            const float* r0 = reinterpret_cast<const float*>(&bReg[0]);
            const float* r1 = reinterpret_cast<const float*>(&bReg[2]);
#pragma unroll
            for (int jj = 0; jj < 8; ++jj) {
                PackU pk;
                pk.h[0] = (_Float16)r0[jj];
                pk.h[1] = (_Float16)r1[jj];
                *reinterpret_cast<unsigned*>(&nxtB[(n8 + jj) * LDS_STRIDE + kp]) = pk.u;
            }
        }
        async_wait0();      // A-tile async copies complete before the barrier
        __syncthreads();
    }

    // ---- epilogue: +bias, relu, store f16.  C layout: VGPR r, lane l ->
    // M = r + 8*(l>=16), N = l&15 within each 16x16 tile.
#pragma unroll
    for (int mi = 0; mi < 4; ++mi) {
#pragma unroll
        for (int ni = 0; ni < 2; ++ni) {
            int n = blockN + wn * 32 + ni * 16 + l15;
            float bv = bias[n];
#pragma unroll
            for (int r = 0; r < 8; ++r) {
                int m = blockM + wm * 64 + mi * 16 + r + hlf * 8;
                float v = acc[mi][ni][r] + bv;
                v = v > 0.f ? v : 0.f;
                C[(size_t)m * N + n] = (_Float16)v;
            }
        }
    }
}

// ---------------------------------------------------------------------------
// Build pair-concat matrix xf (1024 x 3072) f16 from x (4,16,1536) f32.
// Row p=(b,i,j): [x[b,i] ; x[b,j]]
// ---------------------------------------------------------------------------
__global__ void build_xf(const float* __restrict__ x, _Float16* __restrict__ xf)
{
    int p = blockIdx.x;
    int bt = p >> 8, i = (p >> 4) & 15, j = p & 15;
    const float* xi = x + (size_t)(bt * 16 + i) * 1536;
    const float* xj = x + (size_t)(bt * 16 + j) * 1536;
    _Float16* row = xf + (size_t)p * 3072;
    for (int t = threadIdx.x; t < 1536; t += blockDim.x) {
        row[t]        = (_Float16)xi[t];
        row[1536 + t] = (_Float16)xj[t];
    }
}

// ---------------------------------------------------------------------------
// Head layer: out[z,p,0:2] = H3[z,p,:] @ W3[z] + b3[z]   (no activation)
// ---------------------------------------------------------------------------
__global__ void head_gemm(const _Float16* __restrict__ H3, long long strideH,
                          const float* __restrict__ W3, long long strideW,
                          const float* __restrict__ b3, long long strideB,
                          float* __restrict__ out, long long strideO)
{
    int z = blockIdx.y;
    const _Float16* h = H3 + z * strideH;
    const float* w = W3 + z * strideW;
    const float* bb = b3 + z * strideB;
    float* o = out + z * strideO;
    int p = blockIdx.x * blockDim.x + threadIdx.x;
    if (p >= 1024) return;
    const _Float16* hr = h + (size_t)p * 512;
    float a0 = bb[0], a1 = bb[1];
    for (int k = 0; k < 512; ++k) {
        float v = (float)hr[k];
        a0 += v * w[k * 2 + 0];
        a1 += v * w[k * 2 + 1];
    }
    o[p * 2 + 0] = a0;
    o[p * 2 + 1] = a1;
}

// ---------------------------------------------------------------------------
// emo_proj: ef = LN(relu(relu(ep@W0+b0)@W1+b1)) ; one block per (b,n) row.
// ---------------------------------------------------------------------------
__global__ void __launch_bounds__(128)
emo_proj_ln(const float* __restrict__ ep,
            const float* __restrict__ W0, const float* __restrict__ b0,
            const float* __restrict__ W1, const float* __restrict__ b1,
            const float* __restrict__ g,  const float* __restrict__ bln,
            float* __restrict__ ef)
{
    __shared__ float hid[128];
    __shared__ float outv[512];
    __shared__ float red[128];
    int r = blockIdx.x;
    int t = threadIdx.x;
    const float* e = ep + r * 7;
    float acc = b0[t];
    for (int k = 0; k < 7; ++k) acc += e[k] * W0[k * 128 + t];
    hid[t] = acc > 0.f ? acc : 0.f;
    __syncthreads();
    for (int o = t; o < 512; o += 128) {
        float a = b1[o];
        for (int k = 0; k < 128; ++k) a += hid[k] * W1[k * 512 + o];
        outv[o] = a > 0.f ? a : 0.f;
    }
    __syncthreads();
    float s = 0.f;
    for (int o = t; o < 512; o += 128) s += outv[o];
    red[t] = s; __syncthreads();
    for (int st = 64; st > 0; st >>= 1) { if (t < st) red[t] += red[t + st]; __syncthreads(); }
    float mean = red[0] * (1.f / 512.f);
    __syncthreads();
    float vs = 0.f;
    for (int o = t; o < 512; o += 128) { float d = outv[o] - mean; vs += d * d; }
    red[t] = vs; __syncthreads();
    for (int st = 64; st > 0; st >>= 1) { if (t < st) red[t] += red[t + st]; __syncthreads(); }
    float inv = rsqrtf(red[0] * (1.f / 512.f) + 1e-5f);
    __syncthreads();
    for (int o = t; o < 512; o += 128)
        ef[(size_t)r * 512 + o] = (outv[o] - mean) * inv * g[o] + bln[o];
}

// ---------------------------------------------------------------------------
// LN(featH3) + ef  ->  scorer MLP 512->128(relu)->7 -> softmax * pairmask.
// One block (128 thr) per pair row.
// ---------------------------------------------------------------------------
__global__ void __launch_bounds__(128)
scorer_attn(const _Float16* __restrict__ featH3,
            const float* __restrict__ ef,
            const float* __restrict__ ln_g, const float* __restrict__ ln_b,
            const float* __restrict__ Ws0, const float* __restrict__ bs0,
            const float* __restrict__ Ws1, const float* __restrict__ bs1,
            const int* __restrict__ un,
            float* __restrict__ attn)
{
    __shared__ float xrow[512];
    __shared__ float hid[128];
    __shared__ float red[128];
    int p = blockIdx.x;
    int bt = p >> 8, i = (p >> 4) & 15, j = p & 15;
    int t = threadIdx.x;
    const _Float16* hr = featH3 + (size_t)p * 512;
    float s = 0.f;
    for (int o = t; o < 512; o += 128) { float v = (float)hr[o]; xrow[o] = v; s += v; }
    red[t] = s; __syncthreads();
    for (int st = 64; st > 0; st >>= 1) { if (t < st) red[t] += red[t + st]; __syncthreads(); }
    float mean = red[0] * (1.f / 512.f);
    __syncthreads();
    float vs = 0.f;
    for (int o = t; o < 512; o += 128) { float d = xrow[o] - mean; vs += d * d; }
    red[t] = vs; __syncthreads();
    for (int st = 64; st > 0; st >>= 1) { if (t < st) red[t] += red[t + st]; __syncthreads(); }
    float inv = rsqrtf(red[0] * (1.f / 512.f) + 1e-5f);
    __syncthreads();
    const float* efr = ef + (size_t)(bt * 16 + i) * 512;
    for (int o = t; o < 512; o += 128)
        xrow[o] = (xrow[o] - mean) * inv * ln_g[o] + ln_b[o] + efr[o];
    __syncthreads();
    float acc = bs0[t];
    for (int k = 0; k < 512; ++k) acc += xrow[k] * Ws0[k * 128 + t];
    hid[t] = acc > 0.f ? acc : 0.f;
    __syncthreads();
    if (t < 7) {
        float a = bs1[t];
        for (int k = 0; k < 128; ++k) a += hid[k] * Ws1[k * 7 + t];
        red[t] = a;
    }
    __syncthreads();
    if (t == 0) {
        float mx = red[0];
        for (int e2 = 1; e2 < 7; ++e2) mx = fmaxf(mx, red[e2]);
        float ex[7], sum = 0.f;
        for (int e2 = 0; e2 < 7; ++e2) { ex[e2] = __expf(red[e2] - mx); sum += ex[e2]; }
        float pm = ((i < un[bt]) && (j < un[bt])) ? 1.f : 0.f;
        float rs = 1.f / sum;
        for (int e2 = 0; e2 < 7; ++e2) attn[p * 7 + e2] = ex[e2] * rs * pm;
    }
}

// ---------------------------------------------------------------------------
// Final: per-expert softmax gates + top-1, MoE combine, top-3 emotion mix,
// shared mean, residual, neutral term, row mask.
// ---------------------------------------------------------------------------
__global__ void combine(const float* __restrict__ gatesRaw, // (6,1024,2)
                        const float* __restrict__ eo,       // (12,1024,2)
                        const float* __restrict__ sh,       // (2,1024,2)
                        const float* __restrict__ res,      // (1024,2)
                        const float* __restrict__ attn,     // (1024,7)
                        const float* __restrict__ ebias,    // (7,2)
                        const int* __restrict__ un,
                        float* __restrict__ out)            // (1024,2)
{
    int p = blockIdx.x * blockDim.x + threadIdx.x;
    if (p >= 1024) return;
    int bt = p >> 8, i = (p >> 4) & 15;

    float moe0[6], moe1[6];
    for (int e = 0; e < 6; ++e) {
        float g0 = gatesRaw[(size_t)(e * 1024 + p) * 2 + 0];
        float g1 = gatesRaw[(size_t)(e * 1024 + p) * 2 + 1];
        float mx = fmaxf(g0, g1);
        float e0 = __expf(g0 - mx), e1 = __expf(g1 - mx);
        float den = 1.f / (e0 + e1);
        float gs0 = e0 * den, gs1 = e1 * den;
        float bi0 = gs0 + ebias[(1 + e) * 2 + 0];
        float bi1 = gs1 + ebias[(1 + e) * 2 + 1];
        int kidx = (bi0 >= bi1) ? 0 : 1;          // top_k tie -> lower index
        float gsel = (kidx == 0) ? gs0 : gs1;
        float gn = gsel / (gsel + 1e-9f);
        const float* eop = eo + (size_t)((e * 2 + kidx) * 1024 + p) * 2;
        moe0[e] = gn * eop[0];
        moe1[e] = gn * eop[1];
    }

    float awa[6];
    for (int e = 0; e < 6; ++e) awa[e] = attn[p * 7 + 1 + e];
    bool sel[6] = {false, false, false, false, false, false};
    float emo0 = 0.f, emo1 = 0.f;
    for (int t3 = 0; t3 < 3; ++t3) {           // TOPM = 3
        int best = -1; float bv = -1e30f;
        for (int e = 0; e < 6; ++e)
            if (!sel[e] && awa[e] > bv) { bv = awa[e]; best = e; }
        sel[best] = true;
        emo0 += awa[best] * moe0[best];
        emo1 += awa[best] * moe1[best];
    }

    float sh0 = 0.5f * (sh[(size_t)p * 2 + 0] + sh[(size_t)(1024 + p) * 2 + 0]);
    float sh1 = 0.5f * (sh[(size_t)p * 2 + 1] + sh[(size_t)(1024 + p) * 2 + 1]);
    float a0 = attn[p * 7 + 0];
    float l0 = res[p * 2 + 0] + sh0 + emo0 + 0.5f * a0 * sh0;  // NEUTRAL_W=0.5
    float l1 = res[p * 2 + 1] + sh1 + emo1 + 0.5f * a0 * sh1;
    float rm = (i < un[bt]) ? 1.f : 0.f;
    out[p * 2 + 0] = l0 * rm;
    out[p * 2 + 1] = l1 * rm;
}

// ---------------------------------------------------------------------------
// Host: input flattening assumed in setup_inputs() insertion order:
// 0 x, 1 emotion_probs, 2 utterance_nums, then params leaves W,b per layer:
// feat_proj[3..8] emo_proj[9..12] scorer[13..16] ln_g[17] ln_b[18]
// gating[19..26] experts[27..34] shared[35..42] residual[43..50] expert_bias[51]
// ---------------------------------------------------------------------------
extern "C" void kernel_launch(void* const* d_in, const int* in_sizes, int n_in,
                              void* d_out, int out_size, void* d_ws, size_t ws_size,
                              hipStream_t stream)
{
    (void)in_sizes; (void)n_in; (void)out_size;
    const float* x    = (const float*)d_in[0];
    const float* epr  = (const float*)d_in[1];
    const int*   un   = (const int*)d_in[2];
    const float* fpW0 = (const float*)d_in[3];
    const float* fpb0 = (const float*)d_in[4];
    const float* fpW1 = (const float*)d_in[5];
    const float* fpb1 = (const float*)d_in[6];
    const float* fpW2 = (const float*)d_in[7];
    const float* fpb2 = (const float*)d_in[8];
    const float* emW0 = (const float*)d_in[9];
    const float* emb0 = (const float*)d_in[10];
    const float* emW1 = (const float*)d_in[11];
    const float* emb1 = (const float*)d_in[12];
    const float* scW0 = (const float*)d_in[13];
    const float* scb0 = (const float*)d_in[14];
    const float* scW1 = (const float*)d_in[15];
    const float* scb1 = (const float*)d_in[16];
    const float* lng  = (const float*)d_in[17];
    const float* lnb  = (const float*)d_in[18];
    const float* gtW0 = (const float*)d_in[19];
    const float* gtb0 = (const float*)d_in[20];
    const float* gtW1 = (const float*)d_in[21];
    const float* gtb1 = (const float*)d_in[22];
    const float* gtW2 = (const float*)d_in[23];
    const float* gtb2 = (const float*)d_in[24];
    const float* gtW3 = (const float*)d_in[25];
    const float* gtb3 = (const float*)d_in[26];
    const float* exW0 = (const float*)d_in[27];
    const float* exb0 = (const float*)d_in[28];
    const float* exW1 = (const float*)d_in[29];
    const float* exb1 = (const float*)d_in[30];
    const float* exW2 = (const float*)d_in[31];
    const float* exb2 = (const float*)d_in[32];
    const float* exW3 = (const float*)d_in[33];
    const float* exb3 = (const float*)d_in[34];
    const float* shW0 = (const float*)d_in[35];
    const float* shb0 = (const float*)d_in[36];
    const float* shW1 = (const float*)d_in[37];
    const float* shb1 = (const float*)d_in[38];
    const float* shW2 = (const float*)d_in[39];
    const float* shb2 = (const float*)d_in[40];
    const float* shW3 = (const float*)d_in[41];
    const float* shb3 = (const float*)d_in[42];
    const float* rsW0 = (const float*)d_in[43];
    const float* rsb0 = (const float*)d_in[44];
    const float* rsW1 = (const float*)d_in[45];
    const float* rsb1 = (const float*)d_in[46];
    const float* rsW2 = (const float*)d_in[47];
    const float* rsb2 = (const float*)d_in[48];
    const float* rsW3 = (const float*)d_in[49];
    const float* rsb3 = (const float*)d_in[50];
    const float* ebias= (const float*)d_in[51];
    float* out = (float*)d_out;

    // --- workspace layout ---
    char* ws = (char*)d_ws;
    size_t off = 0;
    auto alloc = [&](size_t bytes) -> char* {
        off = (off + 255) & ~(size_t)255;
        char* pp = ws + off;
        off += bytes;
        return pp;
    };
    _Float16* XF      = (_Float16*)alloc((size_t)1024 * 3072 * 2);
    _Float16* featH3  = (_Float16*)alloc((size_t)1024 * 512 * 2);
    float*    gatesRaw= (float*)alloc((size_t)6  * 1024 * 2 * 4);
    float*    eoBuf   = (float*)alloc((size_t)12 * 1024 * 2 * 4);
    float*    shBuf   = (float*)alloc((size_t)2  * 1024 * 2 * 4);
    float*    resBuf  = (float*)alloc((size_t)1024 * 2 * 4);
    float*    efBuf   = (float*)alloc((size_t)64 * 512 * 4);
    float*    attnBuf = (float*)alloc((size_t)1024 * 7 * 4);

    const size_t perStack = ((size_t)1024 * 2048 + (size_t)1024 * 1024 + (size_t)1024 * 512) * 2;
    size_t rem = (ws_size > off + 255) ? (ws_size - off - 255) : 0;
    int zcap = (int)(rem / perStack);
    if (zcap < 1)  zcap = 1;   // assume ws holds at least one stack of scratch
    if (zcap > 12) zcap = 12;
    _Float16* H1 = (_Float16*)alloc((size_t)zcap * 1024 * 2048 * 2);
    _Float16* H2 = (_Float16*)alloc((size_t)zcap * 1024 * 1024 * 2);
    _Float16* H3 = (_Float16*)alloc((size_t)zcap * 1024 * 512 * 2);

    dim3 blk256(256);

    // 1) pair matrix + emotion projection
    build_xf<<<dim3(1024), blk256, 0, stream>>>(x, XF);
    emo_proj_ln<<<dim3(64), dim3(128), 0, stream>>>(epr, emW0, emb0, emW1, emb1, lng, lnb, efBuf);

    // 2) the 22 heavy stacks as WMMA GEMM chains
    auto runChain = [&](int z, const float* W0, const float* b0,
                        const float* W1, const float* b1,
                        const float* W2, const float* b2,
                        _Float16* H3out, long long sH3) {
        gemm_wmma_f16<<<dim3(16, 8, z), blk256, 0, stream>>>(
            XF, 0LL, W0, 3072LL * 2048, b0, 2048LL,
            H1, 1024LL * 2048, 1024, 2048, 3072);
        gemm_wmma_f16<<<dim3(8, 8, z), blk256, 0, stream>>>(
            H1, 1024LL * 2048, W1, 2048LL * 1024, b1, 1024LL,
            H2, 1024LL * 1024, 1024, 1024, 2048);
        gemm_wmma_f16<<<dim3(4, 8, z), blk256, 0, stream>>>(
            H2, 1024LL * 1024, W2, 1024LL * 512, b2, 512LL,
            H3out, sH3, 1024, 512, 1024);
    };

    // feat_proj (H3 persists for scorer)
    runChain(1, fpW0, fpb0, fpW1, fpb1, fpW2, fpb2, featH3, 0LL);
    scorer_attn<<<dim3(1024), dim3(128), 0, stream>>>(
        featH3, efBuf, lng, lnb, scW0, scb0, scW1, scb1, un, attnBuf);

    auto runGroup = [&](int cnt,
                        const float* W0, const float* b0, const float* W1, const float* b1,
                        const float* W2, const float* b2, const float* W3, const float* b3,
                        float* headOut) {
        for (int s0 = 0; s0 < cnt; s0 += zcap) {
            int z = cnt - s0; if (z > zcap) z = zcap;
            runChain(z,
                     W0 + (size_t)s0 * 3072 * 2048, b0 + (size_t)s0 * 2048,
                     W1 + (size_t)s0 * 2048 * 1024, b1 + (size_t)s0 * 1024,
                     W2 + (size_t)s0 * 1024 * 512,  b2 + (size_t)s0 * 512,
                     H3, 1024LL * 512);
            head_gemm<<<dim3(4, z), blk256, 0, stream>>>(
                H3, 1024LL * 512,
                W3 + (size_t)s0 * 512 * 2, 512LL * 2,
                b3 + (size_t)s0 * 2, 2LL,
                headOut + (size_t)s0 * 1024 * 2, 1024LL * 2);
        }
    };

    runGroup(6,  gtW0, gtb0, gtW1, gtb1, gtW2, gtb2, gtW3, gtb3, gatesRaw);
    runGroup(12, exW0, exb0, exW1, exb1, exW2, exb2, exW3, exb3, eoBuf);
    runGroup(2,  shW0, shb0, shW1, shb1, shW2, shb2, shW3, shb3, shBuf);
    runGroup(1,  rsW0, rsb0, rsW1, rsb1, rsW2, rsb2, rsW3, rsb3, resBuf);

    // 3) final gather/combine into d_out
    combine<<<dim3(4), blk256, 0, stream>>>(
        gatesRaw, eoBuf, shBuf, resBuf, attnBuf, ebias, un, out);
}
